// GPT2Block_55576876810791
// MI455X (gfx1250) — matrix operand order repaired and checked by reading, not compiled, not run on hardware
//
#include <hip/hip_runtime.h>
#include <hip/hip_bf16.h>

// ---------------------------------------------------------------------------
// GPT-2 block for MI455X (gfx1250).
//  - All GEMMs: bf16 WMMA, f32 accumulate. Activations staged to LDS by the
//    Tensor Data Mover (tensor_load_to_lds, double buffered, s_wait_tensorcnt).
//    Weights pre-transposed to [N][K] bf16 so WMMA B-fragments are contiguous
//    16B global reads, two per lane (served from the 192MB L2) -- no LDS transpose.
//  - Attention: flash-attention, one wave per 16-query tile, QK^T and PV on
//    WMMA; V is materialized transposed by the QKV epilogue so PV fragments
//    are contiguous global reads.
// ---------------------------------------------------------------------------

typedef __attribute__((ext_vector_type(16))) __bf16 v16bf;
typedef __attribute__((ext_vector_type(8)))  __bf16 v8bf;
typedef __attribute__((ext_vector_type(8)))  float  v8f;
typedef __attribute__((ext_vector_type(4)))  unsigned su4;
typedef __attribute__((ext_vector_type(8)))  unsigned su8;

union V16 { v16bf v; v8bf h[2]; };

static __device__ __forceinline__ __bf16 f2bf(float f) {
    union { float f; unsigned int u; } in; in.f = f;
    unsigned int u = in.u;
    unsigned int r = u + 0x7fffu + ((u >> 16) & 1u);   // round-to-nearest-even
    union { unsigned short s; __bf16 b; } out;
    out.s = (unsigned short)(r >> 16);
    return out.b;
}

static __device__ __forceinline__ v8f wmma_bf16(v16bf a, v16bf b, v8f c) {
    return __builtin_amdgcn_wmma_f32_16x16x32_bf16(
        /*neg_a=*/false, a, /*neg_b=*/false, b,
        /*c_mod=*/(short)0, c, /*reuse_a=*/false, /*reuse_b=*/false);
}

// TDM: 2D tile load, 128 rows x 32 bf16, global row stride = stride_elems,
// LDS rows padded to 80B (pad_interval 16 DWORDs, pad_amount 4 DWORDs).
// D# packing per CDNA5 ISA.
static __device__ __forceinline__ void tdm_load_tile_a(
        unsigned lds_byte_addr, unsigned long long gaddr,
        unsigned tensor_d0, unsigned stride_elems) {
    su4 g0;
    g0[0] = 1u;                                   // count=1, user mode
    g0[1] = lds_byte_addr;                        // lds_addr
    g0[2] = (unsigned)gaddr;                      // global_addr[31:0]
    g0[3] = (unsigned)((gaddr >> 32) & 0x1FFFFFFull) | (2u << 30); // [56:32] | type=2

    su8 g1;
    g1[0] = (1u << 16) | (1u << 20) | (3u << 22) | (3u << 25);
    g1[1] = (tensor_d0 & 0xFFFFu) << 16;          // tensor_dim0[15:0]
    unsigned td1 = 1u << 20;                      // generous tensor_dim1
    g1[2] = (tensor_d0 >> 16) | ((td1 & 0xFFFFu) << 16);
    g1[3] = (td1 >> 16) | (32u << 16);            // tile_dim0 = 32
    g1[4] = 128u;                                 // tile_dim1 = 128
    g1[5] = stride_elems;                         // tensor_dim0_stride
    g1[6] = 0u;
    g1[7] = 0u;
    asm volatile("tensor_load_to_lds %0, %1" :: "s"(g0), "s"(g1) : "memory");
}

static __device__ __forceinline__ void wait_tdm() {
    __builtin_amdgcn_s_wait_tensorcnt(0);
}

// ---------------------------------------------------------------------------
// f32 [K][N] -> bf16 transposed [N][K]  (runs once per weight)
// ---------------------------------------------------------------------------
__global__ __launch_bounds__(256) void k_transpose_bf16(const float* __restrict__ in,
                                                        __bf16* __restrict__ out,
                                                        int K, int N) {
    int i = blockIdx.x * 256 + threadIdx.x;
    if (i < K * N) {
        const int k = i / N, n = i - k * N;
        out[(size_t)n * K + k] = f2bf(in[i]);
    }
}

// ---------------------------------------------------------------------------
// LayerNorm: one block per row, output bf16
// ---------------------------------------------------------------------------
__global__ __launch_bounds__(256) void k_layernorm(const float* __restrict__ x,
                                                   const float* __restrict__ g,
                                                   const float* __restrict__ b,
                                                   __bf16* __restrict__ out, int C) {
    __shared__ float rs[256], rs2[256];
    const int row = blockIdx.x;
    const int tid = threadIdx.x;
    const float* xr = x + (size_t)row * C;
    float s = 0.f, s2 = 0.f;
    for (int c = tid; c < C; c += 256) { float v = xr[c]; s += v; s2 += v * v; }
    rs[tid] = s; rs2[tid] = s2;
    __syncthreads();
    for (int off = 128; off > 0; off >>= 1) {
        if (tid < off) { rs[tid] += rs[tid + off]; rs2[tid] += rs2[tid + off]; }
        __syncthreads();
    }
    const float inv = 1.0f / (float)C;
    const float mu = rs[0] * inv;
    const float var = rs2[0] * inv - mu * mu;
    const float rstd = rsqrtf(var + 1e-5f);
    __bf16* orow = out + (size_t)row * C;
    for (int c = tid; c < C; c += 256)
        orow[c] = f2bf((xr[c] - mu) * rstd * g[c] + b[c]);
}

// ---------------------------------------------------------------------------
// bf16 WMMA GEMM: C[M,N] = A[M,K] * Wt[N,K]^T + bias, fused epilogues.
// 128 threads = 4 waves; block tile 128x128; wave tile 64x64; k-step 32.
// A tile: TDM -> LDS, double buffered. B fragments: direct global reads of Wt.
// ---------------------------------------------------------------------------
#define MODE_QKV  0   // scatter bf16 into q/k [B*H,T,64] and v^T [B*H,64,T]
#define MODE_PROJ 1   // f32 out = acc + bias + res
#define MODE_FF1  2   // bf16 out = gelu(acc + bias)
#define MODE_FF2  3   // f32 out = acc + bias + res

template <int MODE>
__global__ __launch_bounds__(128) void k_gemm_bf16(
    const __bf16* __restrict__ A, const __bf16* __restrict__ Wt,
    const float* __restrict__ bias, const float* __restrict__ res,
    float* __restrict__ outF, __bf16* __restrict__ outB,
    __bf16* __restrict__ qb, __bf16* __restrict__ kb, __bf16* __restrict__ vb,
    int M, int N, int K) {
    (void)M;
    __shared__ __bf16 As[2][128][40];    // A tile (+pad, matches TDM pad)

    const int tid = threadIdx.x;
    const int lane = tid & 31;
    const int wave = tid >> 5;
    const int wm = wave & 1;             // 2 waves along M
    const int wn = wave >> 1;            // 2 waves along N
    const int lrow = lane & 15;
    const int lhalf = lane >> 4;

    const int rowbase = blockIdx.y * 128;
    const int colbase = blockIdx.x * 128;

    v8f acc[4][4];
#pragma unroll
    for (int m = 0; m < 4; ++m)
#pragma unroll
        for (int n = 0; n < 4; ++n) acc[m][n] = (v8f)0.0f;

    const unsigned long long abase =
        (unsigned long long)(uintptr_t)A + (unsigned long long)rowbase * K * 2ull;

    // per-lane base pointers for the 4 B fragments (rows of Wt)
    const __bf16* wrow[4];
#pragma unroll
    for (int n = 0; n < 4; ++n)
        wrow[n] = Wt + (size_t)(colbase + wn * 64 + n * 16 + lrow) * K + lhalf * 8;

    auto stageA = [&](int buf, int k0) {
        if (wave == 0) {
            unsigned lds = (unsigned)(uintptr_t)&As[buf][0][0];
            tdm_load_tile_a(lds, abase + (unsigned)k0 * 2u, (unsigned)K, (unsigned)K);
        }
    };

    // ---- prologue ---------------------------------------------------------
    stageA(0, 0);
    if (wave == 0) wait_tdm();
    __syncthreads();

    // ---- main loop (double buffered A) ------------------------------------
    int buf = 0;
    for (int k0 = 0; k0 < K; k0 += 32, buf ^= 1) {
        const int nk = k0 + 32;
        if (nk < K) stageA(buf ^ 1, nk);

        // B fragments straight from global (L2-resident transposed weights)
        V16 bfr[4];
#pragma unroll
        for (int n = 0; n < 4; ++n) {
            const __bf16* p = wrow[n] + k0;
            bfr[n].h[0] = *(const v8bf*)p;
            bfr[n].h[1] = *(const v8bf*)(p + 16);
        }
        // A fragments from LDS
        V16 af[4];
#pragma unroll
        for (int m = 0; m < 4; ++m) {
            const int r = wm * 64 + m * 16 + lrow;
            const int kboff = lhalf * 8;
            af[m].h[0] = *(const v8bf*)&As[buf][r][kboff];
            af[m].h[1] = *(const v8bf*)&As[buf][r][kboff + 16];
        }
#pragma unroll
        for (int m = 0; m < 4; ++m)
#pragma unroll
            for (int n = 0; n < 4; ++n)
                acc[m][n] = wmma_bf16(af[m].v, bfr[n].v, acc[m][n]);

        if (nk < K && wave == 0) wait_tdm();
        __syncthreads();
    }

    // ---- epilogue ---------------------------------------------------------
#pragma unroll
    for (int m = 0; m < 4; ++m)
#pragma unroll
        for (int n = 0; n < 4; ++n) {
            const int gcol = colbase + wn * 64 + n * 16 + lrow;
            const float bia = bias[gcol];
#pragma unroll
            for (int r = 0; r < 8; ++r) {
                const int grow = rowbase + wm * 64 + m * 16 + r + 8 * lhalf;
                float v = acc[m][n][r] + bia;
                if (MODE == MODE_QKV) {
                    const int which = gcol >> 10;         // 0=q 1=k 2=v
                    const int c = gcol & 1023;
                    const int head = c >> 6, d = c & 63;
                    const int bidx = grow >> 11, tloc = grow & 2047;
                    const size_t bhbase = (size_t)(bidx * 16 + head);
                    if (which == 2) {                     // v stored transposed
                        vb[(bhbase * 64 + d) * 2048 + tloc] = f2bf(v);
                    } else {
                        __bf16* dst = (which == 0) ? qb : kb;
                        dst[(bhbase * 2048 + tloc) * 64 + d] = f2bf(v);
                    }
                } else if (MODE == MODE_PROJ) {
                    outF[(size_t)grow * N + gcol] = v + res[(size_t)grow * N + gcol];
                } else if (MODE == MODE_FF1) {
                    const float ge = 0.5f * v * (1.0f + erff(v * 0.70710678118654752f));
                    outB[(size_t)grow * N + gcol] = f2bf(ge);
                } else { // MODE_FF2
                    outF[(size_t)grow * N + gcol] = v + res[(size_t)grow * N + gcol];
                }
            }
        }
}

// ---------------------------------------------------------------------------
// Flash attention: 128 threads = 4 independent waves, each owns one 16-query
// tile and a private LDS P-slice (no block barrier: trip counts differ).
// q,k: bf16 [B*H, T, 64]; v: bf16 [B*H, 64, T] (transposed at creation).
// ---------------------------------------------------------------------------
__global__ __launch_bounds__(128) void k_attention(const __bf16* __restrict__ Q,
                                                   const __bf16* __restrict__ Kk,
                                                   const __bf16* __restrict__ Vt,
                                                   __bf16* __restrict__ Y) {
    const int T = 2048, HD = 64, C = 1024;
    __shared__ __bf16 Ps[4][16][40];     // per-wave P tile 16x32 (+pad)

    const int wv = threadIdx.x >> 5;
    const int lane = threadIdx.x & 31;
    const int q0 = (blockIdx.x * 4 + wv) * 16;
    const int bh = blockIdx.y;
    const int lrow = lane & 15;
    const int lhalf = lane >> 4;
    const int kboff = lhalf * 8;

    const __bf16* qbase = Q  + (size_t)bh * T * HD;
    const __bf16* kbase = Kk + (size_t)bh * T * HD;
    const __bf16* vbase = Vt + (size_t)bh * HD * T;   // [64][T]

    V16 qa[2];
#pragma unroll
    for (int c = 0; c < 2; ++c) {
        const __bf16* p = qbase + (size_t)(q0 + lrow) * HD + c * 32 + kboff;
        qa[c].h[0] = *(const v8bf*)p;
        qa[c].h[1] = *(const v8bf*)(p + 16);
    }

    v8f o[4];
#pragma unroll
    for (int nt = 0; nt < 4; ++nt) o[nt] = (v8f)0.0f;
    float mi[8], li[8];
#pragma unroll
    for (int r = 0; r < 8; ++r) { mi[r] = -1e30f; li[r] = 0.f; }

    const int kvend = q0 + 16;
    for (int kv0 = 0; kv0 < kvend; kv0 += 32) {
        // ---- S = Q K^T for 16x32 keys
        v8f s[2]; s[0] = (v8f)0.0f; s[1] = (v8f)0.0f;
#pragma unroll
        for (int st = 0; st < 2; ++st) {
#pragma unroll
            for (int c = 0; c < 2; ++c) {
                const __bf16* kp = kbase + (size_t)(kv0 + st * 16 + lrow) * HD
                                   + c * 32 + lhalf * 8;
                V16 kf;
                kf.h[0] = *(const v8bf*)kp;
                kf.h[1] = *(const v8bf*)(kp + 16);
                s[st] = wmma_bf16(qa[c].v, kf.v, s[st]);
            }
        }

        // ---- scale, causal mask, online softmax
        float p0[8], p1[8], alpha[8];
#pragma unroll
        for (int r = 0; r < 8; ++r) {
            const int rowq = q0 + r + 8 * lhalf;
            const int c0 = kv0 + lrow, c1 = c0 + 16;
            float v0 = s[0][r] * 0.125f; if (c0 > rowq) v0 = -1e30f;
            float v1 = s[1][r] * 0.125f; if (c1 > rowq) v1 = -1e30f;
            float mx = fmaxf(v0, v1);
#pragma unroll
            for (int off = 1; off < 16; off <<= 1)
                mx = fmaxf(mx, __shfl_xor(mx, off, 16));
            const float mnew = fmaxf(mi[r], mx);
            alpha[r] = __expf(mi[r] - mnew);
            p0[r] = __expf(v0 - mnew);
            p1[r] = __expf(v1 - mnew);
            float rsum = p0[r] + p1[r];
#pragma unroll
            for (int off = 1; off < 16; off <<= 1)
                rsum += __shfl_xor(rsum, off, 16);
            li[r] = li[r] * alpha[r] + rsum;
            mi[r] = mnew;
        }
#pragma unroll
        for (int nt = 0; nt < 4; ++nt)
#pragma unroll
            for (int r = 0; r < 8; ++r) o[nt][r] *= alpha[r];

        // ---- relayout P (C-layout -> A-layout) through wave-local LDS
#pragma unroll
        for (int r = 0; r < 8; ++r) {
            Ps[wv][r + 8 * lhalf][lrow]      = f2bf(p0[r]);
            Ps[wv][r + 8 * lhalf][lrow + 16] = f2bf(p1[r]);
        }
        __builtin_amdgcn_sched_barrier(0);   // keep LDS store->load order
        V16 pa;
        pa.h[0] = *(const v8bf*)&Ps[wv][lrow][kboff];
        pa.h[1] = *(const v8bf*)&Ps[wv][lrow][kboff + 16];

        // ---- O += P V : B-fragments are contiguous reads of V^T
#pragma unroll
        for (int nt = 0; nt < 4; ++nt) {
            const __bf16* vp = vbase + (size_t)(nt * 16 + lrow) * T + kv0 + lhalf * 8;
            V16 vf;
            vf.h[0] = *(const v8bf*)vp;
            vf.h[1] = *(const v8bf*)(vp + 16);
            o[nt] = wmma_bf16(pa.v, vf.v, o[nt]);
        }
        __builtin_amdgcn_sched_barrier(0);   // before next-iteration LDS stores
    }

    // ---- normalize and write y [B,T,C] (head-interleaved columns), bf16
    const int b = bh >> 4, h = bh & 15;
#pragma unroll
    for (int nt = 0; nt < 4; ++nt)
#pragma unroll
        for (int r = 0; r < 8; ++r) {
            const int t = q0 + r + 8 * lhalf;
            const float val = o[nt][r] / li[r];
            Y[((size_t)(b * T + t)) * C + h * 64 + nt * 16 + lrow] = f2bf(val);
        }
}

// ---------------------------------------------------------------------------
// Host launcher
// ---------------------------------------------------------------------------
extern "C" void kernel_launch(void* const* d_in, const int* in_sizes, int n_in,
                              void* d_out, int out_size, void* d_ws, size_t ws_size,
                              hipStream_t stream) {
    (void)in_sizes; (void)n_in; (void)out_size; (void)ws_size;
    const int B = 2, T = 2048, C = 1024, H3 = 3072, FF = 4096;
    const int M = B * T;   // 4096 token rows

    const float* x      = (const float*)d_in[0];
    const float* ln1_g  = (const float*)d_in[1];
    const float* ln1_b  = (const float*)d_in[2];
    const float* w_attn = (const float*)d_in[3];
    const float* b_attn = (const float*)d_in[4];
    const float* w_proj = (const float*)d_in[5];
    const float* b_proj = (const float*)d_in[6];
    const float* ln2_g  = (const float*)d_in[7];
    const float* ln2_b  = (const float*)d_in[8];
    const float* w_ff1  = (const float*)d_in[9];
    const float* b_ff1  = (const float*)d_in[10];
    const float* w_ff2  = (const float*)d_in[11];
    const float* b_ff2  = (const float*)d_in[12];
    float* out = (float*)d_out;

    size_t off = 0;
    auto alloc = [&](size_t bytes) -> void* {
        void* p = (char*)d_ws + off;
        off += (bytes + 255) & ~(size_t)255;
        return p;
    };
    __bf16* wAttnT = (__bf16*)alloc((size_t)C * H3 * 2);   // [3C][C]
    __bf16* wProjT = (__bf16*)alloc((size_t)C * C  * 2);   // [C][C]
    __bf16* wF1T   = (__bf16*)alloc((size_t)C * FF * 2);   // [4C][C]
    __bf16* wF2T   = (__bf16*)alloc((size_t)FF * C * 2);   // [C][4C]
    __bf16* xn1   = (__bf16*)alloc((size_t)M * C * 2);
    __bf16* qbuf  = (__bf16*)alloc((size_t)M * C * 2);     // [B*H][T][64]
    __bf16* kbuf  = (__bf16*)alloc((size_t)M * C * 2);     // [B*H][T][64]
    __bf16* vbuf  = (__bf16*)alloc((size_t)M * C * 2);     // [B*H][64][T]
    __bf16* yatt  = (__bf16*)alloc((size_t)M * C * 2);
    float*  x1    = (float*) alloc((size_t)M * C * 4);
    __bf16* xn2   = (__bf16*)alloc((size_t)M * C * 2);
    __bf16* hbuf  = (__bf16*)alloc((size_t)M * FF * 2);

    {
        int n;
        n = C * H3; k_transpose_bf16<<<(n + 255) / 256, 256, 0, stream>>>(w_attn, wAttnT, C, H3);
        n = C * C;  k_transpose_bf16<<<(n + 255) / 256, 256, 0, stream>>>(w_proj, wProjT, C, C);
        n = C * FF; k_transpose_bf16<<<(n + 255) / 256, 256, 0, stream>>>(w_ff1,  wF1T,  C, FF);
        n = FF * C; k_transpose_bf16<<<(n + 255) / 256, 256, 0, stream>>>(w_ff2,  wF2T,  FF, C);
    }

    k_layernorm<<<M, 256, 0, stream>>>(x, ln1_g, ln1_b, xn1, C);

    k_gemm_bf16<MODE_QKV><<<dim3(H3 / 128, M / 128), 128, 0, stream>>>(
        xn1, wAttnT, b_attn, nullptr, nullptr, nullptr, qbuf, kbuf, vbuf, M, H3, C);

    k_attention<<<dim3(T / 64, B * 16), 128, 0, stream>>>(qbuf, kbuf, vbuf, yatt);

    k_gemm_bf16<MODE_PROJ><<<dim3(C / 128, M / 128), 128, 0, stream>>>(
        yatt, wProjT, b_proj, x, x1, nullptr, nullptr, nullptr, nullptr, M, C, C);

    k_layernorm<<<M, 256, 0, stream>>>(x1, ln2_g, ln2_b, xn2, C);

    k_gemm_bf16<MODE_FF1><<<dim3(FF / 128, M / 128), 128, 0, stream>>>(
        xn2, wF1T, b_ff1, nullptr, nullptr, hbuf, nullptr, nullptr, nullptr, M, FF, C);

    k_gemm_bf16<MODE_FF2><<<dim3(C / 128, M / 128), 128, 0, stream>>>(
        hbuf, wF2T, b_ff2, x1, out, nullptr, nullptr, nullptr, nullptr, M, C, FF);
}
